// inference_21045339750754
// MI455X (gfx1250) — matrix operand order repaired and checked by reading, not compiled, not run on hardware
//
#include <hip/hip_runtime.h>

#define NN 2048
#define FF 256
#define HH 8
#define DD 64
#define CC 16

typedef __attribute__((ext_vector_type(16))) _Float16 v16h;
typedef __attribute__((ext_vector_type(8)))  _Float16 v8h;
typedef __attribute__((ext_vector_type(8)))  float    v8f;

__device__ __forceinline__ v16h cat8(v8h lo, v8h hi) {
  v16h r;
#pragma unroll
  for (int i = 0; i < 8; ++i) { r[i] = lo[i]; r[8 + i] = hi[i]; }
  return r;
}

__device__ __forceinline__ v8f wmma_f16(v16h a, v16h b, v8f c) {
  // D = A(16x32 f16) * B(32x16 f16) + C(16x16 f32)
  return __builtin_amdgcn_wmma_f32_16x16x32_f16(false, a, false, b, (short)0, c,
                                                false, false);
}

// ---------------------------------------------------------------------------
// Prep: transpose weights into B-fragment-friendly f16 layouts.
// W1 [H][F][D] -> w1t [H][D][F];  W2 [H][H*D][C] -> w2t [H][C][H*D]
// ---------------------------------------------------------------------------
__global__ void k_prep(const float* __restrict__ W1, const float* __restrict__ W2,
                       _Float16* __restrict__ w1t, _Float16* __restrict__ w2t) {
  int i = blockIdx.x * blockDim.x + threadIdx.x;
  if (i < HH * FF * DD) {
    int h = i / (FF * DD), rem = i % (FF * DD);
    int f = rem / DD, d = rem % DD;
    w1t[(h * DD + d) * FF + f] = (_Float16)W1[i];
  }
  if (i < HH * HH * DD * CC) {
    int h = i / (HH * DD * CC), rem = i % (HH * DD * CC);
    int k = rem / CC, c = rem % CC;
    w2t[(h * CC + c) * (HH * DD) + k] = (_Float16)W2[i];
  }
}

// ---------------------------------------------------------------------------
// Layer-1 projection: fts[h] = x @ W1[h]  (K=256 via 8 WMMAs)
// Emits f1 = fts.a1 + b1, f2 = fts.a2 + b2, and fts transposed [h][d][node]
// in f16 (so attention B fragments are two contiguous 8-half runs).
// Block = 128 (4 waves, one 16-col d-tile each); grid = (N/16, H).
// ---------------------------------------------------------------------------
__global__ void k_fts1(const float* __restrict__ x, const _Float16* __restrict__ w1t,
                       const float* __restrict__ a1, const float* __restrict__ b1,
                       const float* __restrict__ a2, const float* __restrict__ b2,
                       _Float16* __restrict__ fts1T, float* __restrict__ f1,
                       float* __restrict__ f2) {
  const int r0 = blockIdx.x * 16;
  const int h  = blockIdx.y;
  const int lane = threadIdx.x & 31;
  const int wv   = threadIdx.x >> 5;
  const int lo = lane & 15, hi = lane >> 4;

  v8f acc = {};
  const float*    xr = x + (size_t)(r0 + lo) * FF;                   // A row M=lo
  const _Float16* bp = w1t + (size_t)(h * DD + wv * 16 + lo) * FF;   // B col N=lo
  for (int k0 = 0; k0 < FF; k0 += 32) {
    v16h a;
#pragma unroll
    for (int i = 0; i < 8; ++i) a[i] = (_Float16)xr[k0 + hi * 8 + i];
#pragma unroll
    for (int i = 0; i < 8; ++i) a[8 + i] = (_Float16)xr[k0 + 16 + hi * 8 + i];
    v16h b = cat8(*(const v8h*)(bp + k0 + hi * 8), *(const v8h*)(bp + k0 + 16 + hi * 8));   // two 8-K runs (table order), 16B aligned
    acc = wmma_f16(a, b, acc);
  }

  __shared__ float fs[16][DD];
#pragma unroll
  for (int r = 0; r < 8; ++r) fs[r + 8 * hi][wv * 16 + lo] = acc[r];
  __syncthreads();

  if (threadIdx.x < 16) {
    int row = threadIdx.x;
    float s1 = 0.f, s2 = 0.f;
    for (int d = 0; d < DD; ++d) {
      float v = fs[row][d];
      s1 += v * a1[h * DD + d];
      s2 += v * a2[h * DD + d];
    }
    f1[h * NN + r0 + row] = s1 + b1[h];
    f2[h * NN + r0 + row] = s2 + b2[h];
  }
  for (int t = threadIdx.x; t < 16 * DD; t += 128) {
    int row = t & 15, d = t >> 4;
    fts1T[(size_t)(h * DD + d) * NN + r0 + row] = (_Float16)fs[row][d];
  }
}

// ---------------------------------------------------------------------------
// Flash GAT attention, layer 1 — register-resident, barrier-free main loop.
// Block = 128 (4 waves); each wave owns a disjoint set of 32-col chunks and
// the full 16x64 output (4 WMMA acc tiles) with its own running (m, l).
// Per lane: compute the 16 score elements of its OWN A-fragment slots
// (row=lane&15; K groups kb=8*(lane>>4) and 16+kb), exp in registers,
// row reductions via __shfl_xor(16) across the lane pair, alpha broadcast to
// C-layout rows via __shfl. LDS used only for the final 4-way softmax merge.
// ---------------------------------------------------------------------------
__global__ void k_attn1(const float* __restrict__ adj, const float* __restrict__ f1,
                        const float* __restrict__ f2, const _Float16* __restrict__ fts1T,
                        const float* __restrict__ bz1, _Float16* __restrict__ h1f) {
  const int r0 = blockIdx.x * 16, h = blockIdx.y;
  const int lane = threadIdx.x & 31, wv = threadIdx.x >> 5;
  const int lo = lane & 15, hi = lane >> 4, kb = hi * 8;

  const float f1s = f1[h * NN + r0 + lo];   // score row term for this lane's row
  const float* f2h = f2 + h * NN;
  const float* ar  = adj + (size_t)(r0 + lo) * NN;

  v8f acc[4] = {};
  float mrun = -3.0e38f, lrun = 0.f;

  for (int it = 0; it < NN / 128; ++it) {     // 16 iterations per wave
    const int c0 = (it * 4 + wv) * 32;
    // ---- scores for this lane's 16 A-fragment elements ----
    float adjv[16], f2v[16];
#pragma unroll
    for (int j = 0; j < 8; ++j) { adjv[j] = ar[c0 + kb + j];      f2v[j] = f2h[c0 + kb + j]; }
#pragma unroll
    for (int j = 0; j < 8; ++j) { adjv[8+j] = ar[c0 + 16 + kb + j]; f2v[8+j] = f2h[c0 + 16 + kb + j]; }
    float s[16];
    float cm = -3.0e38f;
#pragma unroll
    for (int j = 0; j < 16; ++j) {
      float lg = f1s + f2v[j];
      float lr = lg > 0.f ? lg : 0.2f * lg;
      float sv = lr * adjv[j] - 1.0e9f * (1.f - adjv[j]);
      s[j] = sv;
      cm = fmaxf(cm, sv);
    }
    cm = fmaxf(cm, __shfl_xor(cm, 16));       // full-row max (lane pair)
    float mn = fmaxf(mrun, cm);
    float al = __expf(mrun - mn);
    mrun = mn;
    // ---- P = exp(s - m) directly into the A fragment ----
    v16h aF;
    float ls = 0.f;
#pragma unroll
    for (int j = 0; j < 16; ++j) {
      float p = __expf(s[j] - mn);
      ls += p;
      aF[j] = (_Float16)p;
    }
    ls += __shfl_xor(ls, 16);
    lrun = lrun * al + ls;
    // ---- rescale acc by alpha of its C-layout rows (r + 8*hi) ----
    float alr[8];
#pragma unroll
    for (int r = 0; r < 8; ++r) alr[r] = __shfl(al, r + 8 * hi);
#pragma unroll
    for (int t = 0; t < 4; ++t)
#pragma unroll
      for (int r = 0; r < 8; ++r) acc[t][r] *= alr[r];
    // ---- 4 independent WMMAs (full D width) ----
#pragma unroll
    for (int t = 0; t < 4; ++t) {
      v16h b = cat8(*(const v8h*)(fts1T + (size_t)(h * DD + t * 16 + lo) * NN + c0 + hi * 8), *(const v8h*)(fts1T + (size_t)(h * DD + t * 16 + lo) * NN + c0 + 16 + hi * 8));
      acc[t] = wmma_f16(aF, b, acc[t]);
    }
    if (it + 1 < NN / 128)
      __builtin_prefetch(ar + c0 + 128, 0, 3);  // next adj chunk for this wave
  }

  // ---- merge the 4 per-wave partial softmaxes ----
  __shared__ float accS[4][16][DD];
  __shared__ float mS[4][16], lS[4][16], cw[4][16], lstar[16];
  if (hi == 0) { mS[wv][lo] = mrun; lS[wv][lo] = lrun; }
#pragma unroll
  for (int t = 0; t < 4; ++t)
#pragma unroll
    for (int r = 0; r < 8; ++r) accS[wv][r + 8 * hi][t * 16 + lo] = acc[t][r];
  __syncthreads();
  if (threadIdx.x < 16) {
    int row = threadIdx.x;
    float ms = fmaxf(fmaxf(mS[0][row], mS[1][row]), fmaxf(mS[2][row], mS[3][row]));
    float lt = 0.f;
#pragma unroll
    for (int w = 0; w < 4; ++w) {
      float c = __expf(mS[w][row] - ms);
      cw[w][row] = c;
      lt += lS[w][row] * c;
    }
    lstar[row] = lt;
  }
  __syncthreads();
  for (int idx = threadIdx.x; idx < 16 * DD; idx += 128) {
    int row = idx >> 6, d = idx & 63;
    float v = 0.f;
#pragma unroll
    for (int w = 0; w < 4; ++w) v += accS[w][row][d] * cw[w][row];
    v = v / lstar[row] + bz1[h * DD + d];
    v = v > 0.f ? v : (__expf(v) - 1.f);       // ELU
    h1f[(size_t)(r0 + row) * (HH * DD) + h * DD + d] = (_Float16)v;
  }
}

// ---------------------------------------------------------------------------
// Layer-2 projection: fts2[h] = h1 @ W2[h]  (K=512 via 16 WMMAs), 1 wave/blk.
// Emits f1b/f2b and fts2 transposed [h][c][node] f16.
// ---------------------------------------------------------------------------
__global__ void k_fts2(const _Float16* __restrict__ h1f, const _Float16* __restrict__ w2t,
                       const float* __restrict__ c1, const float* __restrict__ d1,
                       const float* __restrict__ c2, const float* __restrict__ d2,
                       _Float16* __restrict__ fts2T, float* __restrict__ f1b,
                       float* __restrict__ f2b) {
  const int r0 = blockIdx.x * 16, h = blockIdx.y;
  const int lane = threadIdx.x & 31;
  const int lo = lane & 15, hi = lane >> 4;

  v8f acc = {};
  const _Float16* ar = h1f + (size_t)(r0 + lo) * (HH * DD);
  const _Float16* bp = w2t + (size_t)(h * CC + lo) * (HH * DD);
  for (int k0 = 0; k0 < HH * DD; k0 += 32) {
    v8h al8 = *(const v8h*)(ar + k0 + hi * 8);
    v8h ah8 = *(const v8h*)(ar + k0 + 16 + hi * 8);
    v16h a = cat8(al8, ah8);
    v16h b = cat8(*(const v8h*)(bp + k0 + hi * 8), *(const v8h*)(bp + k0 + 16 + hi * 8));
    acc = wmma_f16(a, b, acc);
  }

  __shared__ float fs[16][CC + 1];
#pragma unroll
  for (int r = 0; r < 8; ++r) fs[r + 8 * hi][lo] = acc[r];
  __syncthreads();
  if (lane < 16) {
    int row = lane;
    float s1 = 0.f, s2 = 0.f;
    for (int c = 0; c < CC; ++c) {
      float v = fs[row][c];
      s1 += v * c1[h * CC + c];
      s2 += v * c2[h * CC + c];
    }
    f1b[h * NN + r0 + row] = s1 + d1[h];
    f2b[h * NN + r0 + row] = s2 + d2[h];
  }
  __syncthreads();
  for (int t = lane; t < 16 * CC; t += 32) {
    int row = t & 15, c = t >> 4;
    fts2T[(size_t)(h * CC + c) * NN + r0 + row] = (_Float16)fs[row][c];
  }
}

// ---------------------------------------------------------------------------
// Flash attention, layer 2 (C=16, identity act) — same register-resident
// split-column scheme, one acc tile per wave, 4-way merge, out2 [h][n][c].
// ---------------------------------------------------------------------------
__global__ void k_attn2(const float* __restrict__ adj, const float* __restrict__ f1b,
                        const float* __restrict__ f2b, const _Float16* __restrict__ fts2T,
                        const float* __restrict__ bz2, float* __restrict__ out2) {
  const int r0 = blockIdx.x * 16, h = blockIdx.y;
  const int lane = threadIdx.x & 31, wv = threadIdx.x >> 5;
  const int lo = lane & 15, hi = lane >> 4, kb = hi * 8;

  const float f1s = f1b[h * NN + r0 + lo];
  const float* f2h = f2b + h * NN;
  const float* ar  = adj + (size_t)(r0 + lo) * NN;

  v8f acc = {};
  float mrun = -3.0e38f, lrun = 0.f;

  for (int it = 0; it < NN / 128; ++it) {
    const int c0 = (it * 4 + wv) * 32;
    float adjv[16], f2v[16];
#pragma unroll
    for (int j = 0; j < 8; ++j) { adjv[j] = ar[c0 + kb + j];        f2v[j] = f2h[c0 + kb + j]; }
#pragma unroll
    for (int j = 0; j < 8; ++j) { adjv[8+j] = ar[c0 + 16 + kb + j]; f2v[8+j] = f2h[c0 + 16 + kb + j]; }
    float s[16];
    float cm = -3.0e38f;
#pragma unroll
    for (int j = 0; j < 16; ++j) {
      float lg = f1s + f2v[j];
      float lr = lg > 0.f ? lg : 0.2f * lg;
      float sv = lr * adjv[j] - 1.0e9f * (1.f - adjv[j]);
      s[j] = sv;
      cm = fmaxf(cm, sv);
    }
    cm = fmaxf(cm, __shfl_xor(cm, 16));
    float mn = fmaxf(mrun, cm);
    float al = __expf(mrun - mn);
    mrun = mn;
    v16h aF;
    float ls = 0.f;
#pragma unroll
    for (int j = 0; j < 16; ++j) {
      float p = __expf(s[j] - mn);
      ls += p;
      aF[j] = (_Float16)p;
    }
    ls += __shfl_xor(ls, 16);
    lrun = lrun * al + ls;
    float alr[8];
#pragma unroll
    for (int r = 0; r < 8; ++r) alr[r] = __shfl(al, r + 8 * hi);
#pragma unroll
    for (int r = 0; r < 8; ++r) acc[r] *= alr[r];
    v16h b = cat8(*(const v8h*)(fts2T + (size_t)(h * CC + lo) * NN + c0 + hi * 8), *(const v8h*)(fts2T + (size_t)(h * CC + lo) * NN + c0 + 16 + hi * 8));
    acc = wmma_f16(aF, b, acc);
    if (it + 1 < NN / 128)
      __builtin_prefetch(ar + c0 + 128, 0, 3);
  }

  __shared__ float accS[4][16][CC];
  __shared__ float mS[4][16], lS[4][16], cw[4][16], lstar[16];
  if (hi == 0) { mS[wv][lo] = mrun; lS[wv][lo] = lrun; }
#pragma unroll
  for (int r = 0; r < 8; ++r) accS[wv][r + 8 * hi][lo] = acc[r];
  __syncthreads();
  if (threadIdx.x < 16) {
    int row = threadIdx.x;
    float ms = fmaxf(fmaxf(mS[0][row], mS[1][row]), fmaxf(mS[2][row], mS[3][row]));
    float lt = 0.f;
#pragma unroll
    for (int w = 0; w < 4; ++w) {
      float c = __expf(mS[w][row] - ms);
      cw[w][row] = c;
      lt += lS[w][row] * c;
    }
    lstar[row] = lt;
  }
  __syncthreads();
  for (int idx = threadIdx.x; idx < 16 * CC; idx += 128) {
    int row = idx >> 4, c = idx & 15;
    float v = 0.f;
#pragma unroll
    for (int w = 0; w < 4; ++w) v += accS[w][row][c] * cw[w][row];
    v = v / lstar[row] + bz2[h * CC + c];
    out2[((size_t)h * NN + r0 + row) * CC + c] = v;
  }
}

// Mean over heads -> d_out [N][C]
__global__ void k_mean(const float* __restrict__ out2, float* __restrict__ out) {
  int i = blockIdx.x * blockDim.x + threadIdx.x;
  if (i < NN * CC) {
    float s = 0.f;
#pragma unroll
    for (int h = 0; h < HH; ++h) s += out2[(size_t)h * NN * CC + i];
    out[i] = s * (1.0f / HH);
  }
}

extern "C" void kernel_launch(void* const* d_in, const int* in_sizes, int n_in,
                              void* d_out, int out_size, void* d_ws, size_t ws_size,
                              hipStream_t stream) {
  const float* x   = (const float*)d_in[0];
  const float* adj = (const float*)d_in[1];
  // d_in[2] = bias_mat, unused (recomputed from adj: bias = -1e9*(1-adj))
  const float* W1  = (const float*)d_in[3];
  const float* a1  = (const float*)d_in[4];
  const float* b1  = (const float*)d_in[5];
  const float* a2  = (const float*)d_in[6];
  const float* b2  = (const float*)d_in[7];
  const float* bz1 = (const float*)d_in[8];
  const float* W2  = (const float*)d_in[9];
  const float* c1  = (const float*)d_in[10];
  const float* d1  = (const float*)d_in[11];
  const float* c2  = (const float*)d_in[12];
  const float* d2  = (const float*)d_in[13];
  const float* bz2 = (const float*)d_in[14];
  (void)in_sizes; (void)n_in; (void)out_size; (void)ws_size;

  char* ws = (char*)d_ws;
  size_t o = 0;
  auto take = [&](size_t bytes) -> char* {
    char* p = ws + o;
    o = (o + bytes + 255) & ~(size_t)255;
    return p;
  };
  _Float16* w1t   = (_Float16*)take((size_t)HH * DD * FF * 2);        // 256 KB
  _Float16* w2t   = (_Float16*)take((size_t)HH * CC * HH * DD * 2);   // 128 KB
  _Float16* fts1T = (_Float16*)take((size_t)HH * DD * NN * 2);        // 2 MB
  float*    f1    = (float*)take((size_t)HH * NN * 4);
  float*    f2    = (float*)take((size_t)HH * NN * 4);
  _Float16* h1f   = (_Float16*)take((size_t)NN * HH * DD * 2);        // 2 MB
  _Float16* fts2T = (_Float16*)take((size_t)HH * CC * NN * 2);        // 512 KB
  float*    f1b   = (float*)take((size_t)HH * NN * 4);
  float*    f2b   = (float*)take((size_t)HH * NN * 4);
  float*    out2  = (float*)take((size_t)HH * NN * CC * 4);           // 1 MB

  dim3 gtile(NN / 16, HH);
  k_prep<<<dim3((HH * FF * DD + 255) / 256), 256, 0, stream>>>(W1, W2, w1t, w2t);
  k_fts1<<<gtile, 128, 0, stream>>>(x, w1t, a1, b1, a2, b2, fts1T, f1, f2);
  k_attn1<<<gtile, 128, 0, stream>>>(adj, f1, f2, fts1T, bz1, h1f);
  k_fts2<<<gtile, 32, 0, stream>>>(h1f, w2t, c1, d1, c2, d2, fts2T, f1b, f2b);
  k_attn2<<<gtile, 128, 0, stream>>>(adj, f1b, f2b, fts2T, bz2, out2);
  k_mean<<<(NN * CC + 255) / 256, 256, 0, stream>>>(out2, (float*)d_out);
}
